// QuantizedLinear_17042430231143
// MI455X (gfx1250) — matrix operand order repaired and checked by reading, not compiled, not run on hardware
//
#include <hip/hip_runtime.h>

// ---------------------------------------------------------------------------
// AQLM QuantizedLinear on MI455X (gfx1250, wave32).
// Fast path (needs 100MB workspace; A_bf16+W_bf16 = 100MB < 192MB L2):
//   k0: dequantize W -> bf16 [4096,4096]   (one-time)
//   k1: convert A -> bf16 [8192,4096]      (one-time)
//   k2: bf16 WMMA GEMM, 128x256 block tile, 64x64 wave tile, BK=64,
//       tiles staged by the Tensor Data Mover (tensor_load_to_lds) into
//       double-buffered LDS, one barrier per K-step, DMA overlapped with WMMA.
// Fallback: fused dequant+GEMM.
// ---------------------------------------------------------------------------

#define M_TOT 8192
#define N_TOT 4096
#define K_TOT 4096
#define NIG   512
#define NTHR  256

typedef __bf16 bf16_t;
typedef __attribute__((ext_vector_type(16))) __bf16 v16bf;
typedef __attribute__((ext_vector_type(8)))  float  v8f;
typedef __attribute__((ext_vector_type(4)))  unsigned int v4u;
typedef __attribute__((ext_vector_type(8)))  int v8i;
typedef __attribute__((ext_vector_type(4)))  int v4i;

union FragU { uint4 q[2]; v16bf v; };

#if defined(__has_builtin)
#  if __has_builtin(__builtin_amdgcn_tensor_load_to_lds)
#    define HAVE_TDM 1
#  endif
#endif

// ============================ prep kernels =================================

__global__ __launch_bounds__(NTHR)
void dequant_w_bf16(const int*   __restrict__ codes,   // [N,NIG,2]
                    const float* __restrict__ cbooks,  // [2,256,8]
                    const float* __restrict__ scales,  // [N]
                    bf16_t*      __restrict__ Wb)      // [N,K]
{
    const long gid = (long)blockIdx.x * NTHR + threadIdx.x;   // 0 .. N*NIG-1
    const int  n   = (int)(gid >> 9);
    const int2 c   = *(const int2*)(codes + gid * 2);
    const float s  = scales[n];
    const float* e0 = cbooks + c.x * 8;
    const float* e1 = cbooks + 2048 + c.y * 8;
    union { bf16_t h[8]; uint4 u; } pk;
    #pragma unroll
    for (int j = 0; j < 8; ++j)
        pk.h[j] = (bf16_t)((e0[j] + e1[j]) * s);
    *(uint4*)(Wb + gid * 8) = pk.u;
}

__global__ __launch_bounds__(NTHR)
void cvt_a_bf16(const float* __restrict__ A, bf16_t* __restrict__ Ab)
{
    const long base = ((long)blockIdx.x * NTHR + threadIdx.x) * 8;
    const float4 f0 = *(const float4*)(A + base);
    const float4 f1 = *(const float4*)(A + base + 4);
    union { bf16_t h[8]; uint4 u; } pk;
    pk.h[0] = (bf16_t)f0.x; pk.h[1] = (bf16_t)f0.y;
    pk.h[2] = (bf16_t)f0.z; pk.h[3] = (bf16_t)f0.w;
    pk.h[4] = (bf16_t)f1.x; pk.h[5] = (bf16_t)f1.y;
    pk.h[6] = (bf16_t)f1.z; pk.h[7] = (bf16_t)f1.w;
    *(uint4*)(Ab + base) = pk.u;
}

// ============================ GEMM config ==================================

#define BM  128
#define BNF 256
#define BK  64
#define LDA 72          // padded LDS stride (bf16): 64 data + 8 pad = 144B row
#define LDB 72
#define ASZ (BM  * LDA * 2)   // 18432 B per A buffer
#define BSZ (BNF * LDB * 2)   // 36864 B per B buffer

// ============================ TDM GEMM =====================================
#ifdef HAVE_TDM

// 2-D tile load via Tensor Data Mover. data_size=2B; tile = tile_rows x 64;
// tensor row stride = K_TOT; LDS pad: +4 DWORDs (16B) after every 32 DWORDs
// (128B = one 64-elem bf16 row) -> reproduces the 72-element padded stride.
__device__ __forceinline__ void tdm_load_tile(unsigned lds_off, const bf16_t* gsrc,
                                              int tile_rows, int tensor_rows)
{
    const unsigned long long ga = (unsigned long long)(const void*)gsrc;
    v4u g0;
    g0.x = 1u;                                   // count=1 (valid user D#)
    g0.y = lds_off;                              // lds_addr (bytes)
    g0.z = (unsigned)ga;                         // global_addr[31:0]
    g0.w = (unsigned)((ga >> 32) & 0x01ffffffu)  // global_addr[56:32]
         | 0x80000000u;                          // type=2 ("image")
    v8i g1;
    g1[0] = (int)((1u << 16)      // data_size = 2 bytes
                | (1u << 20)      // pad_enable
                | (4u << 22)      // pad_interval: 2^(4+1)=32 DWORDs
                | (3u << 25));    // pad_amount: 4 DWORDs
    g1[1] = (int)((K_TOT & 0xffffu) << 16);            // tensor_dim0[15:0]
    g1[2] = (int)(((unsigned)tensor_rows & 0xffffu) << 16); // td0[31:16]=0 | td1[15:0]
    g1[3] = (int)(64u << 16);                          // td1[31:16]=0 | tile_dim0=64
    g1[4] = tile_rows;                                 // tile_dim1 (tile_dim2=0)
    g1[5] = (int)K_TOT;                                // tensor_dim0_stride[31:0]
    g1[6] = 0;
    g1[7] = 0;
    v4i zz4  = {0, 0, 0, 0};
    v8i zz8  = {0, 0, 0, 0, 0, 0, 0, 0};
    // amdgpu-toolchain (clang-23) 6-arg form:
    // (v4u g0, v8i g1, v4i g2, v4i g3, v8i g4, i32 cpol)
    __builtin_amdgcn_tensor_load_to_lds(g0, g1, zz4, zz4, zz8, 0);
}

__global__ __launch_bounds__(NTHR)
void gemm_bf16_wmma_tdm(const bf16_t* __restrict__ Ab,    // [M,K]
                        const bf16_t* __restrict__ Wb,    // [N,K]
                        const float*  __restrict__ bias,  // [N]
                        float*        __restrict__ out)   // [M,N]
{
    extern __shared__ char smem[];                 // 2*ASZ + 2*BSZ = 110592 B
    bf16_t* const As0 = (bf16_t*)(smem);
    bf16_t* const As1 = (bf16_t*)(smem + ASZ);
    bf16_t* const Bs0 = (bf16_t*)(smem + 2 * ASZ);
    bf16_t* const Bs1 = (bf16_t*)(smem + 2 * ASZ + BSZ);

    const int tid  = threadIdx.x;
    const int lane = tid & 31;
    const int wave = tid >> 5;
    const int wm   = wave >> 2;        // 0..1 : 64-row M slab
    const int wn   = wave & 3;         // 0..3 : 64-col N slab
    const int fl   = lane & 15;
    const int fh   = lane >> 4;

    const int nBase = blockIdx.x * BNF;
    const int mBase = blockIdx.y * BM;

    const bf16_t* aSrc = Ab + (long)mBase * K_TOT;
    const bf16_t* bSrc = Wb + (long)nBase * K_TOT;
    const unsigned asOff0 = (unsigned)(size_t)(void*)As0;
    const unsigned asOff1 = (unsigned)(size_t)(void*)As1;
    const unsigned bsOff0 = (unsigned)(size_t)(void*)Bs0;
    const unsigned bsOff1 = (unsigned)(size_t)(void*)Bs1;

    v8f acc[4][4];
    #pragma unroll
    for (int mt = 0; mt < 4; ++mt)
        #pragma unroll
        for (int nt = 0; nt < 4; ++nt) {
            v8f z = {0.f,0.f,0.f,0.f,0.f,0.f,0.f,0.f};
            acc[mt][nt] = z;
        }

    // preload tile 0 into buffer 0 (TDM ignores EXEC; gate per wave)
    if (wave == 0) tdm_load_tile(asOff0, aSrc, BM,  M_TOT);
    if (wave == 1) tdm_load_tile(bsOff0, bSrc, BNF, N_TOT);

    const int NT = K_TOT / BK;     // 64 K-steps
    for (int t = 0; t < NT; ++t) {
        if (wave < 2) __builtin_amdgcn_s_wait_tensorcnt(0);
        __syncthreads();           // tile t visible; prior reads of buf[(t+1)&1] done

        // overlap: DMA tile t+1 into the other buffer while computing tile t
        if (t + 1 < NT) {
            const int kb1 = (t + 1) * BK;
            if (wave == 0) tdm_load_tile((t & 1) ? asOff0 : asOff1, aSrc + kb1, BM,  M_TOT);
            if (wave == 1) tdm_load_tile((t & 1) ? bsOff0 : bsOff1, bSrc + kb1, BNF, N_TOT);
        }

        const bf16_t* As = (t & 1) ? As1 : As0;
        const bf16_t* Bs = (t & 1) ? Bs1 : Bs0;

        #pragma unroll
        for (int kk = 0; kk < BK; kk += 32) {
            FragU afrag[4], bfrag[4];
            // A 16x32 bf16: lanes 0-15 K[0..7]+[16..23], lanes 16-31 K[8..15]+[24..31]
            #pragma unroll
            for (int mt = 0; mt < 4; ++mt) {
                const bf16_t* p = &As[(wm * 64 + mt * 16 + fl) * LDA + kk + fh * 8];
                afrag[mt].q[0] = *(const uint4*)p;
                afrag[mt].q[1] = *(const uint4*)(p + 16);
            }
            // B 32x16 bf16: lanes 0-15 K[0..7]+[16..23], lanes 16-31 K[8..15]+[24..31]
            #pragma unroll
            for (int nt = 0; nt < 4; ++nt) {
                const bf16_t* p = &Bs[(wn * 64 + nt * 16 + fl) * LDB + kk + fh * 8];
                bfrag[nt].q[0] = *(const uint4*)p;
                bfrag[nt].q[1] = *(const uint4*)(p + 16);
            }
            #pragma unroll
            for (int mt = 0; mt < 4; ++mt)
                #pragma unroll
                for (int nt = 0; nt < 4; ++nt)
                    acc[mt][nt] = __builtin_amdgcn_wmma_f32_16x16x32_bf16(
                        false, afrag[mt].v, false, bfrag[nt].v,
                        (short)0, acc[mt][nt], false, false);
        }
    }

    const int colBase = nBase + wn * 64 + fl;
    float bv[4];
    #pragma unroll
    for (int nt = 0; nt < 4; ++nt) bv[nt] = bias[colBase + nt * 16];
    #pragma unroll
    for (int mt = 0; mt < 4; ++mt) {
        const int mrow = mBase + wm * 64 + mt * 16 + fh * 8;
        #pragma unroll
        for (int nt = 0; nt < 4; ++nt) {
            const int col = colBase + nt * 16;
            #pragma unroll
            for (int r = 0; r < 8; ++r)
                out[(long)(mrow + r) * N_TOT + col] = acc[mt][nt][r] + bv[nt];
        }
    }
}
#endif // HAVE_TDM

// ================== manual-staging GEMM (proven, non-TDM) ==================

__global__ __launch_bounds__(NTHR)
void gemm_bf16_wmma(const bf16_t* __restrict__ Ab,
                    const bf16_t* __restrict__ Wb,
                    const float*  __restrict__ bias,
                    float*        __restrict__ out)
{
    __shared__ bf16_t As[BM  * LDA];
    __shared__ bf16_t Bs[BNF * LDB];

    const int tid  = threadIdx.x;
    const int lane = tid & 31;
    const int wave = tid >> 5;
    const int wm   = wave >> 2;
    const int wn   = wave & 3;
    const int fl   = lane & 15;
    const int fh   = lane >> 4;

    const int nBase = blockIdx.x * BNF;
    const int mBase = blockIdx.y * BM;

    v8f acc[4][4];
    #pragma unroll
    for (int mt = 0; mt < 4; ++mt)
        #pragma unroll
        for (int nt = 0; nt < 4; ++nt) {
            v8f z = {0.f,0.f,0.f,0.f,0.f,0.f,0.f,0.f};
            acc[mt][nt] = z;
        }

    for (int kb = 0; kb < K_TOT; kb += BK) {
        #pragma unroll
        for (int i = 0; i < 4; ++i) {
            const int v   = tid + i * NTHR;
            const int row = v >> 3;
            const int c8  = v & 7;
            const uint4 d = *(const uint4*)(Ab + (long)(mBase + row) * K_TOT + kb + c8 * 8);
            *(uint4*)&As[row * LDA + c8 * 8] = d;
        }
        #pragma unroll
        for (int i = 0; i < 8; ++i) {
            const int v   = tid + i * NTHR;
            const int row = v >> 3;
            const int c8  = v & 7;
            const uint4 d = *(const uint4*)(Wb + (long)(nBase + row) * K_TOT + kb + c8 * 8);
            *(uint4*)&Bs[row * LDB + c8 * 8] = d;
        }
        __syncthreads();

        #pragma unroll
        for (int kk = 0; kk < BK; kk += 32) {
            FragU afrag[4], bfrag[4];
            #pragma unroll
            for (int mt = 0; mt < 4; ++mt) {
                const bf16_t* p = &As[(wm * 64 + mt * 16 + fl) * LDA + kk + fh * 8];
                afrag[mt].q[0] = *(const uint4*)p;
                afrag[mt].q[1] = *(const uint4*)(p + 16);
            }
            #pragma unroll
            for (int nt = 0; nt < 4; ++nt) {
                const bf16_t* p = &Bs[(wn * 64 + nt * 16 + fl) * LDB + kk + fh * 8];
                bfrag[nt].q[0] = *(const uint4*)p;
                bfrag[nt].q[1] = *(const uint4*)(p + 16);
            }
            #pragma unroll
            for (int mt = 0; mt < 4; ++mt)
                #pragma unroll
                for (int nt = 0; nt < 4; ++nt)
                    acc[mt][nt] = __builtin_amdgcn_wmma_f32_16x16x32_bf16(
                        false, afrag[mt].v, false, bfrag[nt].v,
                        (short)0, acc[mt][nt], false, false);
        }
        __syncthreads();
    }

    const int colBase = nBase + wn * 64 + fl;
    float bv[4];
    #pragma unroll
    for (int nt = 0; nt < 4; ++nt) bv[nt] = bias[colBase + nt * 16];
    #pragma unroll
    for (int mt = 0; mt < 4; ++mt) {
        const int mrow = mBase + wm * 64 + mt * 16 + fh * 8;
        #pragma unroll
        for (int nt = 0; nt < 4; ++nt) {
            const int col = colBase + nt * 16;
            #pragma unroll
            for (int r = 0; r < 8; ++r)
                out[(long)(mrow + r) * N_TOT + col] = acc[mt][nt][r] + bv[nt];
        }
    }
}

// ============================ fallback (fused) =============================

#define FBN 128
#define FBK 64

__global__ __launch_bounds__(NTHR)
void aqlm_fused_wmma(const float* __restrict__ A,
                     const int*   __restrict__ codes,
                     const float* __restrict__ cbooks,
                     const float* __restrict__ scales,
                     const float* __restrict__ bias,
                     float* __restrict__ out)
{
    __shared__ bf16_t As[BM * LDA];
    __shared__ bf16_t Bs[FBN * LDB];
    __shared__ float  Cb[2 * 256 * 8];

    const int tid  = threadIdx.x;
    const int lane = tid & 31;
    const int wave = tid >> 5;
    const int wm   = wave >> 2;
    const int wn   = wave & 3;
    const int fl   = lane & 15;
    const int fh   = lane >> 4;

    const int nBase = blockIdx.x * FBN;
    const int mBase = blockIdx.y * BM;

    for (int i = tid; i < 2 * 256 * 8; i += NTHR) Cb[i] = cbooks[i];

    const int   rq    = tid >> 1;
    const int   gq0   = (tid & 1) * 4;
    const float sc    = scales[nBase + rq];
    const long  cbase = (long)(nBase + rq) * (NIG * 2);

    v8f acc[4][2];
    #pragma unroll
    for (int mt = 0; mt < 4; ++mt)
        #pragma unroll
        for (int nt = 0; nt < 2; ++nt) {
            v8f z = {0.f,0.f,0.f,0.f,0.f,0.f,0.f,0.f};
            acc[mt][nt] = z;
        }

    __syncthreads();

    for (int kb = 0; kb < K_TOT; kb += FBK) {
        #pragma unroll
        for (int t = 0; t < 8; ++t) {
            const int v   = tid + t * NTHR;
            const int row = v >> 4;
            const int c4  = v & 15;
            const float4 f = *(const float4*)(A + (long)(mBase + row) * K_TOT + kb + c4 * 4);
            union { bf16_t h[4]; uint2 u; } pk;
            pk.h[0] = (bf16_t)f.x; pk.h[1] = (bf16_t)f.y;
            pk.h[2] = (bf16_t)f.z; pk.h[3] = (bf16_t)f.w;
            *(uint2*)&As[row * LDA + c4 * 4] = pk.u;
        }
        {
            const int gk0 = kb >> 3;
            #pragma unroll
            for (int i = 0; i < 4; ++i) {
                const int g = gq0 + i;
                const int2 c = *(const int2*)(codes + cbase + (long)(gk0 + g) * 2);
                const float* e0 = &Cb[c.x * 8];
                const float* e1 = &Cb[2048 + c.y * 8];
                union { bf16_t h[8]; uint4 u; } pk;
                #pragma unroll
                for (int j = 0; j < 8; ++j)
                    pk.h[j] = (bf16_t)((e0[j] + e1[j]) * sc);
                *(uint4*)&Bs[rq * LDB + g * 8] = pk.u;
            }
        }
        __syncthreads();

        #pragma unroll
        for (int kk = 0; kk < FBK; kk += 32) {
            FragU afrag[4], bfrag[2];
            #pragma unroll
            for (int mt = 0; mt < 4; ++mt) {
                const bf16_t* p = &As[(wm * 64 + mt * 16 + fl) * LDA + kk + fh * 8];
                afrag[mt].q[0] = *(const uint4*)p;
                afrag[mt].q[1] = *(const uint4*)(p + 16);
            }
            #pragma unroll
            for (int nt = 0; nt < 2; ++nt) {
                const bf16_t* p = &Bs[(wn * 32 + nt * 16 + fl) * LDB + kk + fh * 8];
                bfrag[nt].q[0] = *(const uint4*)p;
                bfrag[nt].q[1] = *(const uint4*)(p + 16);
            }
            #pragma unroll
            for (int mt = 0; mt < 4; ++mt)
                #pragma unroll
                for (int nt = 0; nt < 2; ++nt)
                    acc[mt][nt] = __builtin_amdgcn_wmma_f32_16x16x32_bf16(
                        false, afrag[mt].v, false, bfrag[nt].v,
                        (short)0, acc[mt][nt], false, false);
        }
        __syncthreads();
    }

    const int   colBase = nBase + wn * 32 + fl;
    const float b0 = bias[colBase];
    const float b1 = bias[colBase + 16];
    #pragma unroll
    for (int mt = 0; mt < 4; ++mt) {
        const int mrow = mBase + wm * 64 + mt * 16 + fh * 8;
        #pragma unroll
        for (int nt = 0; nt < 2; ++nt) {
            const int   col = colBase + nt * 16;
            const float bb  = nt ? b1 : b0;
            #pragma unroll
            for (int r = 0; r < 8; ++r)
                out[(long)(mrow + r) * N_TOT + col] = acc[mt][nt][r] + bb;
        }
    }
}

// ============================ launcher =====================================

extern "C" void kernel_launch(void* const* d_in, const int* in_sizes, int n_in,
                              void* d_out, int out_size, void* d_ws, size_t ws_size,
                              hipStream_t stream) {
    (void)in_sizes; (void)n_in; (void)out_size;
    const float* A      = (const float*)d_in[0];
    const int*   codes  = (const int*)  d_in[1];
    const float* cbooks = (const float*)d_in[2];
    const float* scales = (const float*)d_in[3];
    const float* bias   = (const float*)d_in[4];
    float*       out    = (float*)d_out;

    const size_t needW = (size_t)N_TOT * K_TOT * sizeof(bf16_t);   // 33.5 MB
    const size_t needA = (size_t)M_TOT * K_TOT * sizeof(bf16_t);   // 67.1 MB

    if (ws_size >= needW + needA) {
        bf16_t* Wb = (bf16_t*)d_ws;
        bf16_t* Ab = (bf16_t*)((char*)d_ws + needW);

        dequant_w_bf16<<<(N_TOT * NIG) / NTHR, NTHR, 0, stream>>>(codes, cbooks, scales, Wb);
        cvt_a_bf16<<<((long)M_TOT * K_TOT) / (NTHR * 8), NTHR, 0, stream>>>(A, Ab);

        dim3 grid(N_TOT / BNF, M_TOT / BM);    // 16 x 64 = 1024 workgroups
#ifdef HAVE_TDM
        gemm_bf16_wmma_tdm<<<grid, NTHR, 2 * ASZ + 2 * BSZ, stream>>>(Ab, Wb, bias, out);
#else
        gemm_bf16_wmma<<<grid, NTHR, 0, stream>>>(Ab, Wb, bias, out);
#endif
    } else {
        dim3 grid(N_TOT / FBN, M_TOT / BM);    // 32 x 64 = 2048 workgroups
        aqlm_fused_wmma<<<grid, NTHR, 0, stream>>>(A, codes, cbooks, scales, bias, out);
    }
}
